// WaveRNN_23373212025087
// MI455X (gfx1250) — matrix operand order repaired and checked by reading, not compiled, not run on hardware
//
#include <hip/hip_runtime.h>

// ---------------------------------------------------------------------------
// WaveRNN on MI455X (gfx1250): persistent WMMA GRU + WMMA projections.
//   inputs (all f32): x(32,256,4096) c(32,80,16) W_ih(2688,336) W_hh(2688,896)
//                     b_ih(2688) b_hh(2688) W2(256,896) b2(256)
//   output f32: (32,256,4096)
//
// Structure:
//  k_pack_xin : lerp-upsample cond + concat + f16 pack, K padded 336->352
//  k_pack_w2  : W2 -> f16, pre-swizzled into WMMA-B lane layout
//  k_gru      : persistent kernel, 112 WGs; each owns 16 batch x 16 hid cols,
//               weights (3 gates, f16, B-swizzled) resident in LDS (117KB).
//               4 waves/WG: waves 0-2 = recurrent h@W_hh per gate (28 WMMA),
//               wave 3 = NEXT step's x@W_ih for all gates (33 WMMA) into a
//               ping-pong LDS stage (x-projection is off the recurrence, so
//               it overlaps; critical chain drops 39->33 WMMAs, and in CU
//               mode the 2-SIMD pairing drops 78->61).
//               1 atomic grid barrier/step; h ping-pongs in f16; f32 h in regs.
//               hidden history stored [b][t][h] for coalesced k_out stores.
//  k_out      : relu(h) @ W2^T + b2; per wave: 1 A tile x 4 B tiles (N=64),
//               each lane stores 8 consecutive-t floats as one 32B v8f.
// ---------------------------------------------------------------------------

typedef _Float16 v16h __attribute__((ext_vector_type(16)));
typedef _Float16 v8h  __attribute__((ext_vector_type(8)));
typedef float    v8f  __attribute__((ext_vector_type(8)));

#define HID    896
#define DIN    336
#define DPAD   352      // 336 padded to 11*32
#define BATCH  32
#define TLEN   4096
#define OUTN   256
#define KT_IH  11       // 352/32
#define KT_HH  28       // 896/32
#define KT_TOT 39
#define NWG    112      // 56 column tiles x 2 batch row tiles
#define GRUTHR 128      // 4 waves: hh-r | hh-z | hh-n | x-stage(t+1)

// ---- workspace layout (bytes) ----
#define OFF_SYNC   0u
#define OFF_H16    256u                      // 2 * 32*896 halfs = 114688 B (ping-pong)
#define OFF_W2P    114944u                   // 16*28*512 halfs  = 458752 B
#define OFF_XIN    573696u                   // 4096*32*352 halfs = 92274688 B
#define OFF_HRELU  92848384u                 // 32*4096*896 halfs = 234881024 B
// total ≈ 327.7 MB

// ---------------------------------------------------------------------------
// WMMA helpers. Operand register layouts used here (wave32); A/B k order is contiguous, not the ISA table's:
//  A 16x32 f16 : lane l holds row m=l%16; halfs 0-15 = K kb+(l<16?0:16)+i
//                -> 16 contiguous halfs (two 16B loads).
//  B 32x16 f16 : lane l holds col n=l%16, K = (l<16?0:16)+i -> we pre-swizzle
//                weights so each lane's 16 halfs are contiguous (32B).
//  C/D 16x16 f32: vgpr v, lane l -> M = v+(l>=16?8:0), N = l%16.
// ---------------------------------------------------------------------------
__device__ __forceinline__ v8f wmma_f16(v16h a, v16h b, v8f c) {
  return __builtin_amdgcn_wmma_f32_16x16x32_f16(false, a, false, b, (short)0, c,
                                                false, false);
}
__device__ __forceinline__ v16h ldA16(const _Float16* p) {  // p = row + kb + (l<16?0:16)
  v8h lo = *(const v8h*)(p);
  v8h hi = *(const v8h*)(p + 8);
  return __builtin_shufflevector(lo, hi, 0,1,2,3,4,5,6,7,8,9,10,11,12,13,14,15);
}
__device__ __forceinline__ v16h ldB16(const _Float16* p) {  // 16 contiguous halfs
  v8h lo = *(const v8h*)(p);
  v8h hi = *(const v8h*)(p + 8);
  return __builtin_shufflevector(lo, hi, 0,1,2,3,4,5,6,7,8,9,10,11,12,13,14,15);
}

// ---------------------------------------------------------------------------
__global__ void k_init(unsigned* __restrict__ p, int n) {
  for (int i = threadIdx.x; i < n; i += blockDim.x) p[i] = 0u;
}

// Build xin f16: [t][b][d] with d<256 = x, 256<=d<336 = lerp-upsampled c, pad 0.
__global__ void k_pack_xin(const float* __restrict__ x, const float* __restrict__ c,
                           _Float16* __restrict__ xin) {
  int idx = blockIdx.x * blockDim.x + threadIdx.x;
  if (idx >= TLEN * BATCH * DPAD) return;
  int t = idx / (BATCH * DPAD);
  int rem = idx - t * (BATCH * DPAD);
  int b = rem / DPAD;
  int d = rem - b * DPAD;
  float v = 0.0f;
  if (d < 256) {
    v = x[((size_t)b * 256 + d) * TLEN + t];
  } else if (d < DIN) {
    int ci = d - 256;
    int cur = t >> 8;                 // t / HOP(256); max 15
    int prev = (cur + 15) & 15;       // (cur-1) mod 16, -1 wraps to last frame
    float w = 1.0f - (float)(t & 255) * (1.0f / 256.0f);
    float cp = c[((size_t)b * 80 + ci) * 16 + prev];
    float cc = c[((size_t)b * 80 + ci) * 16 + cur];
    v = cp + w * (cc - cp);
  }
  xin[((size_t)t * BATCH + b) * DPAD + d] = (_Float16)v;
}

// Pre-swizzle W2 (OUTN x HID, row-major f32) into WMMA-B lane layout, f16.
__global__ void k_pack_w2(const float* __restrict__ W2, _Float16* __restrict__ w2p) {
  int idx = blockIdx.x * blockDim.x + threadIdx.x;
  if (idx >= 16 * KT_HH * 512) return;
  int ntile = idx / (KT_HH * 512);
  int rem = idx - ntile * (KT_HH * 512);
  int kt = rem >> 9;
  int r2 = rem & 511;
  int l = r2 >> 4;
  int i = r2 & 15;
  int K = kt * 32 + ((l < 16) ? 0 : 16) + i;
  int N = ntile * 16 + (l & 15);
  w2p[idx] = (_Float16)W2[(size_t)N * HID + K];
}

// ---------------------------------------------------------------------------
// Persistent GRU. Requires all 112 WGs co-resident (LDS ~126KB < 320KB/WGP).
// ---------------------------------------------------------------------------
__global__ void __launch_bounds__(GRUTHR, 1)
k_gru(const _Float16* __restrict__ xin, _Float16* __restrict__ h16,
      const float* __restrict__ W_ih, const float* __restrict__ W_hh,
      const float* __restrict__ b_ih, const float* __restrict__ b_hh,
      unsigned* __restrict__ syncp, _Float16* __restrict__ hrelu) {
  __shared__ _Float16 sW[3 * KT_TOT * 512];  // 117 KB: [gate][ktile][lane][16]
  __shared__ float sAcc[3 * 256];            // hh staging: r | z | n (+b_hh_n)
  __shared__ float sX[2 * 3 * 256];          // x-projection ping-pong (+biases)

  const int tid = threadIdx.x;
  const int w = tid >> 5;            // waves 0-2: hh gate w; wave 3: x-stage
  const int l = tid & 31;
  const int rt = blockIdx.x & 1;     // batch row tile
  const int ncol = (blockIdx.x >> 1) * 16;  // hidden column base

  // One-time: load + convert + swizzle this WG's weight slice into LDS.
  for (int idx = tid; idx < 3 * KT_TOT * 512; idx += GRUTHR) {
    int g = idx / (KT_TOT * 512);
    int rem = idx - g * (KT_TOT * 512);
    int kt = rem >> 9;
    int r2 = rem & 511;
    int ll = r2 >> 4;
    int ii = r2 & 15;
    int row = g * HID + ncol + (ll & 15);
    float v;
    if (kt < KT_IH) {
      int K = kt * 32 + ((ll < 16) ? 0 : 16) + ii;
      v = (K < DIN) ? W_ih[(size_t)row * DIN + K] : 0.0f;
    } else {
      int K = (kt - KT_IH) * 32 + ((ll < 16) ? 0 : 16) + ii;
      v = W_hh[(size_t)row * HID + K];
    }
    sW[idx] = (_Float16)v;
  }

  float hreg[2] = {0.f, 0.f};        // each thread owns <=2 h elements (f32)
  const int rowm = rt * 16 + (l & 15);
  const int aoff = (l < 16) ? 0 : 16;
  const int crow = (l >= 16) ? 8 : 0;
  const int cn = l & 15;

  // Per-lane bias constants (column-dependent only), folded into stages.
  float biasx[3];                    // wave 3: per-gate additive constants
  biasx[0] = b_ih[ncol + cn] + b_hh[ncol + cn];                   // r
  biasx[1] = b_ih[HID + ncol + cn] + b_hh[HID + ncol + cn];       // z
  biasx[2] = b_ih[2 * HID + ncol + cn];                           // xn
  const float bias_hn = b_hh[2 * HID + ncol + cn];                // hn (wave 2)

  // x-projection of step tt into sX[tt&1] (wave 3 only).
  auto xproj = [&](int tt) {
    float* xs = &sX[(tt & 1) * 768];
    const _Float16* xr = xin + ((size_t)tt * BATCH + rowm) * DPAD + aoff;
    #pragma unroll
    for (int g3 = 0; g3 < 3; ++g3) {
      v8f a = {};
      const _Float16* bB = &sW[(size_t)(g3 * KT_TOT) * 512 + l * 16];
      #pragma unroll
      for (int kt = 0; kt < KT_IH; ++kt)
        a = wmma_f16(ldA16(xr + kt * 32), ldB16(bB + kt * 512), a);
      float bias = biasx[g3];
      #pragma unroll
      for (int v = 0; v < 8; ++v)
        xs[g3 * 256 + (v + crow) * 16 + cn] = a[v] + bias;
    }
  };

  if (w == 3) xproj(0);              // prologue: stage x-projection for t=0
  __syncthreads();

  const _Float16* sBhh = &sW[(size_t)(w * KT_TOT + KT_IH) * 512 + l * 16];

  for (int t = 0; t < TLEN; ++t) {
    _Float16* hwrite = h16 + (size_t)((t & 1) ^ 1) * (BATCH * HID);

    if (w < 3) {
      // Recurrent part: h(t-1) @ W_hh for gate w (28 WMMAs).
      const _Float16* hread =
          h16 + (size_t)(t & 1) * (BATCH * HID) + (size_t)rowm * HID + aoff;
      v8f acc = {};
      #pragma unroll 4
      for (int kt = 0; kt < KT_HH; ++kt)
        acc = wmma_f16(ldA16(hread + kt * 32), ldB16(sBhh + kt * 512), acc);
      float bf = (w == 2) ? bias_hn : 0.f;
      #pragma unroll
      for (int v = 0; v < 8; ++v)
        sAcc[w * 256 + (v + crow) * 16 + cn] = acc[v] + bf;
    } else {
      // Stage next step's input projection (33 WMMAs, off the recurrence).
      if (t + 1 < TLEN) xproj(t + 1);
      // Prefetch step t+2's xin slab (this WG's 16 rows, 11264B).
      if (t + 2 < TLEN) {
        const char* pf =
            (const char*)(xin + ((size_t)(t + 2) * BATCH + rt * 16) * DPAD) + l * 352;
        __builtin_prefetch(pf, 0, 1);
      }
    }
    __syncthreads();

    // Gate nonlinearity + state update for this WG's 256 elements.
    const float* xs = &sX[(t & 1) * 768];
    int ri = 0;
    for (int i = tid; i < 256; i += GRUTHR, ++ri) {
      int m = i >> 4, j = i & 15;
      int hc = ncol + j;
      int b = rt * 16 + m;
      float rv = 1.0f / (1.0f + __expf(-(xs[i] + sAcc[i])));
      float zv = 1.0f / (1.0f + __expf(-(xs[256 + i] + sAcc[256 + i])));
      float nv = tanhf(xs[512 + i] + rv * sAcc[512 + i]);
      float hv = (1.0f - zv) * nv + zv * hreg[ri];
      hreg[ri] = hv;
      hwrite[(size_t)b * HID + hc] = (_Float16)hv;
      // history stored [b][t][h] so k_out tiles span consecutive t
      hrelu[((size_t)b * TLEN + t) * HID + hc] = (_Float16)(hv > 0.f ? hv : 0.f);
    }
    __threadfence();
    __syncthreads();
    if (tid == 0) {
      atomicAdd(syncp, 1u);
      unsigned target = (unsigned)(t + 1) * (unsigned)gridDim.x;
      volatile unsigned* vc = (volatile unsigned*)syncp;
      while (*vc < target) __builtin_amdgcn_s_sleep(1);
    }
    __syncthreads();
  }
}

// ---------------------------------------------------------------------------
// out[b][o][t] = relu(h[t][b]) @ W2^T + b2.  M=(b*4096+t)=131072, N=256, K=896.
// Per wave: one 16-row A tile (fixed b, 16 consecutive t) x 4 B tiles (N=64),
// A kept in registers across the 4 accumulators (4x less A traffic).
// Each lane's accumulator column is 8 consecutive t -> one 32B v8f store.
// ---------------------------------------------------------------------------
__global__ void __launch_bounds__(128)
k_out(const _Float16* __restrict__ hrelu, const _Float16* __restrict__ w2p,
      const float* __restrict__ b2, float* __restrict__ out) {
  int wid = blockIdx.x * 4 + (threadIdx.x >> 5);
  int l = threadIdx.x & 31;
  int mtile = wid >> 2;           // 0..8191 ; row = b*4096 + t
  int og = wid & 3;               // group of 4 otiles -> o base = og*64
  int b = mtile >> 8;             // 4096/16 = 256 tiles per batch row
  int tbase = (mtile & 255) * 16;

  const _Float16* arow =
      hrelu + ((size_t)mtile * 16 + (l & 15)) * HID + ((l < 16) ? 0 : 16);
  const _Float16* bp0 = w2p + ((size_t)(og * 4) * KT_HH) * 512 + l * 16;

  v8f acc[4] = {{}, {}, {}, {}};
  for (int kt = 0; kt < KT_HH; ++kt) {
    v16h a = ldA16(arow + kt * 32);
    #pragma unroll
    for (int j = 0; j < 4; ++j)
      acc[j] = wmma_f16(a, ldB16(bp0 + (size_t)(j * KT_HH + kt) * 512), acc[j]);
  }

  const int crow = (l >= 16) ? 8 : 0;
  #pragma unroll
  for (int j = 0; j < 4; ++j) {
    int o = og * 64 + j * 16 + (l & 15);
    float bias = b2[o];
    v8f ov;
    #pragma unroll
    for (int v = 0; v < 8; ++v) ov[v] = acc[j][v] + bias;
    // acc[j][v] is (t = tbase + crow + v, o): contiguous along t, 32B aligned
    *(v8f*)(out + ((size_t)b * OUTN + o) * TLEN + tbase + crow) = ov;
  }
}

// ---------------------------------------------------------------------------
extern "C" void kernel_launch(void* const* d_in, const int* in_sizes, int n_in,
                              void* d_out, int out_size, void* d_ws, size_t ws_size,
                              hipStream_t stream) {
  const float* x    = (const float*)d_in[0];
  const float* c    = (const float*)d_in[1];
  const float* W_ih = (const float*)d_in[2];
  const float* W_hh = (const float*)d_in[3];
  const float* b_ih = (const float*)d_in[4];
  const float* b_hh = (const float*)d_in[5];
  const float* W2   = (const float*)d_in[6];
  const float* b2   = (const float*)d_in[7];

  char* ws = (char*)d_ws;
  unsigned* syncp   = (unsigned*)(ws + OFF_SYNC);
  _Float16* h16     = (_Float16*)(ws + OFF_H16);
  _Float16* w2p     = (_Float16*)(ws + OFF_W2P);
  _Float16* xin     = (_Float16*)(ws + OFF_XIN);
  _Float16* hrelu   = (_Float16*)(ws + OFF_HRELU);

  // Zero sync counter + both h ping-pong buffers (deterministic replay).
  k_init<<<1, 256, 0, stream>>>((unsigned*)ws, (int)((OFF_H16 + 2u * BATCH * HID * 2u) / 4u));

  int nx = TLEN * BATCH * DPAD;
  k_pack_xin<<<(nx + 255) / 256, 256, 0, stream>>>(x, c, xin);
  k_pack_w2<<<(16 * KT_HH * 512 + 255) / 256, 256, 0, stream>>>(W2, w2p);

  k_gru<<<NWG, GRUTHR, 0, stream>>>(xin, h16, W_ih, W_hh, b_ih, b_hh, syncp, hrelu);

  // 8192 mtiles * 4 otile-groups = 32768 waves = 8192 blocks of 128
  k_out<<<8192, 128, 0, stream>>>(hrelu, w2p, b2, (float*)d_out);
}
